// SafeMoE_64673617543278
// MI455X (gfx1250) — matrix operand order repaired and checked by reading, not compiled, not run on hardware
//
#include <hip/hip_runtime.h>
#include <cmath>

// ---------------------------------------------------------------------------
// Types for CDNA5 WMMA (gfx1250, wave32)
// ---------------------------------------------------------------------------
typedef __attribute__((ext_vector_type(16))) __bf16   v16bf;
typedef __attribute__((ext_vector_type(8)))  float    v8f;
typedef __attribute__((ext_vector_type(4)))  unsigned u32x4;

union Frag {
  v16bf v;
  u32x4 q[2];
};

__device__ __forceinline__ unsigned short f2bf(float f) {
  union { float f; unsigned u; } cv; cv.f = f;
  unsigned u = cv.u;
  u += 0x7FFFu + ((u >> 16) & 1u);   // round-to-nearest-even
  return (unsigned short)(u >> 16);
}

// ---------------------------------------------------------------------------
// 0) zero-init output (we scatter with atomics)
// ---------------------------------------------------------------------------
__global__ void moe_zero_kernel(float* __restrict__ y, size_t n) {
  size_t i = (size_t)blockIdx.x * blockDim.x + threadIdx.x;
  if (i < n) y[i] = 0.f;
}

// ---------------------------------------------------------------------------
// 0b) f32 -> bf16 bulk convert (4 elems/thread, vectorized)
// ---------------------------------------------------------------------------
__global__ void f32_to_bf16_kernel(const float* __restrict__ src,
                                   unsigned short* __restrict__ dst, size_t n4) {
  size_t i = (size_t)blockIdx.x * blockDim.x + threadIdx.x;
  if (i < n4) {
    const float4 v = ((const float4*)src)[i];
    const unsigned lo = (unsigned)f2bf(v.x) | ((unsigned)f2bf(v.y) << 16);
    const unsigned hi = (unsigned)f2bf(v.z) | ((unsigned)f2bf(v.w) << 16);
    ((uint2*)dst)[i] = make_uint2(lo, hi);
  }
}

// ---------------------------------------------------------------------------
// 0c) f32 [K,N] -> bf16 [N,K] transpose-convert (32x32 LDS tiles, batched)
// ---------------------------------------------------------------------------
__global__ __launch_bounds__(256) void transpose_f32_bf16_kernel(
    const float* __restrict__ W, unsigned short* __restrict__ WT,
    int K, int N, size_t strideIn, size_t strideOut) {
  __shared__ unsigned short tile[32][33];
  const float* Wb = W + (size_t)blockIdx.z * strideIn;
  unsigned short* WTb = WT + (size_t)blockIdx.z * strideOut;
  const int k0 = blockIdx.y * 32, n0 = blockIdx.x * 32;
  const int tx = threadIdx.x & 31, ty = threadIdx.x >> 5;   // 32 x 8
#pragma unroll
  for (int r = 0; r < 32; r += 8)
    tile[r + ty][tx] = f2bf(Wb[(size_t)(k0 + r + ty) * N + n0 + tx]);
  __syncthreads();
#pragma unroll
  for (int r = 0; r < 32; r += 8)
    WTb[(size_t)(n0 + r + ty) * K + k0 + tx] = tile[tx][r + ty];
}

// ---------------------------------------------------------------------------
// 1) Router: logits = x @ Wr^T, softmax over E=8, top-2, renormalize.
// ---------------------------------------------------------------------------
__global__ void moe_router_kernel(const float* __restrict__ x,
                                  const float* __restrict__ Wr,
                                  int* __restrict__ t2id, float* __restrict__ t2w,
                                  int T, int D) {
  const int lane = threadIdx.x & 31;
  const int wv   = threadIdx.x >> 5;
  const int t    = blockIdx.x * 8 + wv;
  if (t >= T) return;

  float acc[8];
#pragma unroll
  for (int e = 0; e < 8; ++e) acc[e] = 0.f;
  const float* xr = x + (size_t)t * D;
  for (int k = lane; k < D; k += 32) {
    const float xv = xr[k];
#pragma unroll
    for (int e = 0; e < 8; ++e) acc[e] += xv * Wr[e * D + k];
  }
#pragma unroll
  for (int e = 0; e < 8; ++e)
#pragma unroll
    for (int off = 16; off > 0; off >>= 1)
      acc[e] += __shfl_xor(acc[e], off, 32);

  if (lane == 0) {
    float sc[8], m = -1e30f;
#pragma unroll
    for (int e = 0; e < 8; ++e) { sc[e] = acc[e] * 1.0f /*SCORE_SCALE*/; m = fmaxf(m, sc[e]); }
    float p[8], s = 0.f;
#pragma unroll
    for (int e = 0; e < 8; ++e) { p[e] = expf(sc[e] - m); s += p[e]; }
#pragma unroll
    for (int e = 0; e < 8; ++e) p[e] /= s;
    int e0 = 0;
#pragma unroll
    for (int e = 1; e < 8; ++e) if (p[e] > p[e0]) e0 = e;
    int e1 = (e0 == 0) ? 1 : 0;
#pragma unroll
    for (int e = 0; e < 8; ++e) if (e != e0 && p[e] > p[e1]) e1 = e;
    const float s0 = p[e0], s1 = p[e1];
    const float den = fmaxf(s0 + s1, 1e-9f);
    t2id[2 * t] = e0; t2id[2 * t + 1] = e1;
    t2w[2 * t] = s0 / den; t2w[2 * t + 1] = s1 / den;
  }
}

// ---------------------------------------------------------------------------
// 2) Sequential capacity assignment (exact reference semantics).
// ---------------------------------------------------------------------------
#define MAXT 8192
__global__ void moe_assign_kernel(const int* __restrict__ t2id,
                                  const float* __restrict__ t2w,
                                  int* __restrict__ lists, float* __restrict__ wlists,
                                  int* __restrict__ counts,
                                  int* __restrict__ fblist, int* __restrict__ fbcnt,
                                  int* __restrict__ fbflag,
                                  int T, int cap) {
  __shared__ unsigned char fb[MAXT];
  __shared__ int scan[1024];
  const int tid = threadIdx.x;
  const int tpt = T / 1024;          // tokens per thread (8 for T=8192)

  for (int i = tid; i < T; i += 1024) fb[i] = 0;
  if (tid == 0) *fbcnt = 0;
  __syncthreads();

  for (int e = 0; e < 8; ++e) {
    int elig[8];
    int cnt = 0;
    for (int i = 0; i < tpt; ++i) {
      const int t = tid * tpt + i;
      const bool match = (t2id[2 * t] == e) || (t2id[2 * t + 1] == e);
      const bool el = match && !fb[t];
      elig[i] = el ? 1 : 0;
      cnt += elig[i];
    }
    scan[tid] = cnt;
    __syncthreads();
    for (int off = 1; off < 1024; off <<= 1) {
      const int v = (tid >= off) ? scan[tid - off] : 0;
      __syncthreads();
      scan[tid] += v;
      __syncthreads();
    }
    int pos = scan[tid] - cnt;        // exclusive prefix in token order
    const int total = scan[1023];
    if (tid == 0) counts[e] = (total < cap) ? total : cap;
    for (int i = 0; i < tpt; ++i) {
      if (elig[i]) {
        const int t = tid * tpt + i;
        if (pos < cap) {
          lists[e * cap + pos] = t;
          wlists[e * cap + pos] = (t2id[2 * t] == e) ? t2w[2 * t] : t2w[2 * t + 1];
        } else {
          fb[t] = 1;
          const int fi = atomicAdd(fbcnt, 1);
          fblist[fi] = t;
        }
        ++pos;
      }
    }
    __syncthreads();
  }
  for (int i = tid; i < T; i += 1024) fbflag[i] = fb[i];
}

// ---------------------------------------------------------------------------
// Unified tiled BF16 WMMA GEMM. Block = 256 threads = 8 wave32s.
// Tile 128x128, K-step 64, LDS double-buffered; each wave: 4x2 accumulators,
// 16 v_wmma per K-step. A and B are pre-converted bf16; B pre-transposed to
// [N][K] so both tiles stage as contiguous b128 copies (no VALU in hot loop).
//
// MODE 0: H[slot] = gelu(A[rowlist[slot]] @ B + bias)          (A = xbf)
// MODE 1: Y[tok] += w[slot] * (A[slot] @ B + bias), skip fb    (A = H)
// MODE 2: Y[tok]  = A[slot] @ B + bias                         (A = H)
// ---------------------------------------------------------------------------
#define TM 128
#define TN 128
#define TK 64
#define LDT 72      // 144B rows, 16B-aligned fragment chunks

template <int MODE>
__global__ __launch_bounds__(256) void moe_gemm_kernel(
    const unsigned short* __restrict__ Abf, int K,
    const int* __restrict__ rowlist, const int* __restrict__ countp,
    const unsigned short* __restrict__ BT,   // [N][K] bf16
    const float* __restrict__ bias, int N,
    unsigned short* __restrict__ Hout,
    const float* __restrict__ wlist, const int* __restrict__ fbflag,
    float* __restrict__ Y) {
  __shared__ __align__(16) unsigned short As[2][TM * LDT];
  __shared__ __align__(16) unsigned short Bs[2][TN * LDT];
  const int count = *countp;
  const int mbase = blockIdx.y * TM;
  if (mbase >= count) return;
  const int nbase = blockIdx.x * TN;
  const int tid = threadIdx.x, lane = tid & 31, wv = tid >> 5;
  const int mw = (wv & 1) * 64, nw = (wv >> 1) * 32;

  // staging: thread owns half (64B) of one A row and one B column
  const int arow = tid >> 1, ahalf = (tid & 1) * 32;
  const int aslot = mbase + arow;
  const unsigned short* aptr = nullptr;
  if (aslot < count)
    aptr = Abf + (size_t)((MODE == 0) ? rowlist[aslot] : aslot) * K;
  const unsigned short* bptr = BT + (size_t)(nbase + arow) * K;

  u32x4 ra[4], rb[4];
  auto ldtile = [&](int kb) {
    if (aptr) {
      const u32x4* pa = (const u32x4*)(aptr + kb + ahalf);
      ra[0] = pa[0]; ra[1] = pa[1]; ra[2] = pa[2]; ra[3] = pa[3];
    } else {
      u32x4 z = {};
      ra[0] = z; ra[1] = z; ra[2] = z; ra[3] = z;
    }
    const u32x4* pb = (const u32x4*)(bptr + kb + ahalf);
    rb[0] = pb[0]; rb[1] = pb[1]; rb[2] = pb[2]; rb[3] = pb[3];
  };
  auto sttile = [&](int buf) {
    u32x4* da = (u32x4*)&As[buf][arow * LDT + ahalf];
    da[0] = ra[0]; da[1] = ra[1]; da[2] = ra[2]; da[3] = ra[3];
    u32x4* db = (u32x4*)&Bs[buf][arow * LDT + ahalf];
    db[0] = rb[0]; db[1] = rb[1]; db[2] = rb[2]; db[3] = rb[3];
  };

  v8f zero = {};
  v8f acc[4][2];
#pragma unroll
  for (int m = 0; m < 4; ++m)
#pragma unroll
    for (int n = 0; n < 2; ++n) acc[m][n] = zero;

  ldtile(0);
  sttile(0);
  __syncthreads();

  const int nk = K / TK;
  for (int kt = 0; kt < nk; ++kt) {
    const int cur = kt & 1;
    if (kt + 1 < nk) ldtile((kt + 1) * TK);   // prefetch next tile into regs
#pragma unroll
    for (int s = 0; s < 2; ++s) {             // two 32-wide K slices
      Frag a[4], b[2];
#pragma unroll
      for (int m = 0; m < 4; ++m) {
        const unsigned short* base = &As[cur][(mw + m * 16 + (lane & 15)) * LDT + s * 32];
        a[m].q[0] = *(const u32x4*)(base + ((lane >> 4) * 8));
        a[m].q[1] = *(const u32x4*)(base + 16 + ((lane >> 4) * 8));
      }
#pragma unroll
      for (int n = 0; n < 2; ++n) {
        const unsigned short* p = &Bs[cur][(nw + n * 16 + (lane & 15)) * LDT +
                                               s * 32 + ((lane >> 4) * 8)];
        b[n].q[0] = *(const u32x4*)(p);
        b[n].q[1] = *(const u32x4*)(p + 16);
      }
#pragma unroll
      for (int m = 0; m < 4; ++m)
#pragma unroll
        for (int n = 0; n < 2; ++n)
          acc[m][n] = __builtin_amdgcn_wmma_f32_16x16x32_bf16(
              false, a[m].v, false, b[n].v, (short)0, acc[m][n], false, false);
    }
    if (kt + 1 < nk) sttile(cur ^ 1);         // fill other buffer
    __syncthreads();
  }

  // epilogue: C/D layout = lanes 0-15: VGPR r -> M=r, N=lane; lanes 16-31: M=8+r
  const int rb8 = (lane >> 4) * 8, cc = lane & 15;
#pragma unroll
  for (int m = 0; m < 4; ++m)
#pragma unroll
    for (int n = 0; n < 2; ++n) {
      const int colg = nbase + nw + n * 16 + cc;
      const float bv = bias[colg];
#pragma unroll
      for (int r = 0; r < 8; ++r) {
        const int slot = mbase + mw + m * 16 + rb8 + r;
        if (slot < count) {
          float v = acc[m][n][r] + bv;
          if (MODE == 0) {
            v = 0.5f * v * (1.f + erff(v * 0.70710678118f));   // exact GELU
            Hout[(size_t)slot * N + colg] = f2bf(v);
          } else {
            const int tok = rowlist[slot];
            if (MODE == 2) {
              Y[(size_t)tok * N + colg] = v;                   // fallback overwrite
            } else if (!fbflag[tok]) {
              atomicAdd(&Y[(size_t)tok * N + colg], wlist[slot] * v);
            }
          }
        }
      }
    }
}

// ---------------------------------------------------------------------------
// Host launcher
// ---------------------------------------------------------------------------
extern "C" void kernel_launch(void* const* d_in, const int* in_sizes, int n_in,
                              void* d_out, int out_size, void* d_ws, size_t ws_size,
                              hipStream_t stream) {
  const float* x   = (const float*)d_in[0];
  const float* Wr  = (const float*)d_in[1];
  const float* W1  = (const float*)d_in[2];
  const float* b1  = (const float*)d_in[3];
  const float* W2  = (const float*)d_in[4];
  const float* b2  = (const float*)d_in[5];
  const float* fw1 = (const float*)d_in[6];
  const float* fb1 = (const float*)d_in[7];
  const float* fw2 = (const float*)d_in[8];
  const float* fb2 = (const float*)d_in[9];
  float* y = (float*)d_out;
  (void)n_in; (void)out_size; (void)ws_size;

  const int E = 8;
  const int D = in_sizes[1] / E;          // 1024
  const int T = in_sizes[0] / D;          // 8192
  const int I = in_sizes[3] / E;          // 4096
  int cap = (int)(1.25 * ((double)T * 2.0 / (double)E));
  if (cap < 4) cap = 4;                   // 2560

  // workspace carve-up (256B aligned slices)
  char* ws = (char*)d_ws;
  size_t off = 0;
  auto take = [&](size_t bytes) {
    char* p = ws + off;
    off += (bytes + 255) & ~(size_t)255;
    return p;
  };
  int*   t2id   = (int*)  take((size_t)T * 2 * sizeof(int));
  float* t2w    = (float*)take((size_t)T * 2 * sizeof(float));
  int*   lists  = (int*)  take((size_t)E * cap * sizeof(int));
  float* wlists = (float*)take((size_t)E * cap * sizeof(float));
  int*   counts = (int*)  take((size_t)(E + 1) * sizeof(int));
  int*   fbcnt  = counts + E;
  int*   fblist = (int*)  take((size_t)T * sizeof(int));
  int*   fbflag = (int*)  take((size_t)T * sizeof(int));
  unsigned short* H    = (unsigned short*)take((size_t)T * I * 2);
  unsigned short* xbf  = (unsigned short*)take((size_t)T * D * 2);
  unsigned short* W1T  = (unsigned short*)take((size_t)E * D * I * 2);
  unsigned short* W2T  = (unsigned short*)take((size_t)E * I * D * 2);
  unsigned short* fw1T = (unsigned short*)take((size_t)D * I * 2);
  unsigned short* fw2T = (unsigned short*)take((size_t)I * D * 2);

  const size_t ny = (size_t)T * D;
  moe_zero_kernel<<<dim3((unsigned)((ny + 255) / 256)), 256, 0, stream>>>(y, ny);

  // precision prep: x -> bf16; weights -> bf16 transposed [N][K]
  const size_t nx4 = ((size_t)T * D) / 4;
  f32_to_bf16_kernel<<<dim3((unsigned)((nx4 + 255) / 256)), 256, 0, stream>>>(x, xbf, nx4);
  transpose_f32_bf16_kernel<<<dim3(I / 32, D / 32, E), 256, 0, stream>>>(
      W1, W1T, D, I, (size_t)D * I, (size_t)D * I);
  transpose_f32_bf16_kernel<<<dim3(D / 32, I / 32, E), 256, 0, stream>>>(
      W2, W2T, I, D, (size_t)I * D, (size_t)I * D);
  transpose_f32_bf16_kernel<<<dim3(I / 32, D / 32, 1), 256, 0, stream>>>(
      fw1, fw1T, D, I, 0, 0);
  transpose_f32_bf16_kernel<<<dim3(D / 32, I / 32, 1), 256, 0, stream>>>(
      fw2, fw2T, I, D, 0, 0);

  moe_router_kernel<<<dim3((T + 7) / 8), 256, 0, stream>>>(x, Wr, t2id, t2w, T, D);
  moe_assign_kernel<<<1, 1024, 0, stream>>>(t2id, t2w, lists, wlists, counts,
                                            fblist, fbcnt, fbflag, T, cap);

  const int mtC = (cap + TM - 1) / TM;    // 20
  const int mtT = (T + TM - 1) / TM;      // 64 (worst-case fallback)
  const int ntI = I / TN;                 // 32
  const int ntD = D / TN;                 // 8

  for (int e = 0; e < E; ++e) {
    moe_gemm_kernel<0><<<dim3(ntI, mtC), 256, 0, stream>>>(
        xbf, D, lists + (size_t)e * cap, counts + e,
        W1T + (size_t)e * D * I, b1 + (size_t)e * I, I,
        H, nullptr, nullptr, nullptr);
    moe_gemm_kernel<1><<<dim3(ntD, mtC), 256, 0, stream>>>(
        H, I, lists + (size_t)e * cap, counts + e,
        W2T + (size_t)e * I * D, b2 + (size_t)e * D, D,
        nullptr, wlists + (size_t)e * cap, fbflag, y);
  }
  // fallback FFN for overflow tokens only
  moe_gemm_kernel<0><<<dim3(ntI, mtT), 256, 0, stream>>>(
      xbf, D, fblist, fbcnt, fw1T, fb1, I, H, nullptr, nullptr, nullptr);
  moe_gemm_kernel<2><<<dim3(ntD, mtT), 256, 0, stream>>>(
      H, I, fblist, fbcnt, fw2T, fb2, D, nullptr, nullptr, nullptr, y);
}
